// GraniteMoeMoE_47536698032450
// MI455X (gfx1250) — matrix operand order repaired and checked by reading, not compiled, not run on hardware
//
#include <hip/hip_runtime.h>
#include <hip/hip_bf16.h>

#define T 4096
#define D 2048
#define F 1024
#define E 8
#define TM 16            // token rows per block tile
#define BN 64            // output-column chunk (4 waves x 16)
#define KK 32            // WMMA K for bf16
#define MAXTILE (T / TM) // 256 tiles per expert worst case
#define TILE_ELEMS 2048  // one packed 32K x 64N bf16 tile
#define TILE_BYTES 4096

typedef __attribute__((ext_vector_type(16))) __bf16 v16bf;
typedef __attribute__((ext_vector_type(8)))  __bf16 v8bf;
typedef __attribute__((ext_vector_type(8)))  float  v8f;
typedef __attribute__((ext_vector_type(4)))  unsigned int u32x4;
typedef __attribute__((ext_vector_type(8)))  int i32x8;
typedef __attribute__((ext_vector_type(4)))  int i32x4;

// Dynamic-LDS carve (bytes)
#define LDS_B1   0        // 2 x 4KB double-buffered B tile (w1 / w2)
#define LDS_B2   8192     // 2 x 4KB double-buffered B tile (w3)
#define LDS_ACT  16384    // 32KB packed activation (A-operand chunks)
#define LDS_TOK  49152    // 16 ints
#define LDS_WT   49216    // 16 floats
#define LDS_TOTAL 49280

static __device__ __forceinline__ unsigned short f2bf(float f) {
    unsigned int u = __float_as_uint(f);
    u += 0x7FFFu + ((u >> 16) & 1u);   // round-to-nearest-even
    return (unsigned short)(u >> 16);
}

static __device__ __forceinline__ v8f wmma_bf16(v16bf a, v16bf b, v8f c) {
    return __builtin_amdgcn_wmma_f32_16x16x32_bf16(
        false, a, false, b, (short)0, c, false, false);
}

// TDM: DMA one contiguous 4KB packed tile (2048 bf16) from global into LDS.
// 1-D tensor descriptor per CDNA5 ISA:
//   group0: count=1 | lds_addr | global_addr | type=2
//   group1: data_size=2B, tensor_dim0=tile_dim0=2048, tensor_dim1=1, stride=2048
// This toolchain's builtin is the 6-arg form; trailing groups are unused (zero).
static __device__ __forceinline__ void tdm_load_tile(unsigned lds_byte_off,
                                                     const unsigned short* gsrc) {
    const unsigned long long ga = (unsigned long long)(size_t)gsrc;
    u32x4 g0;
    g0[0] = 1u;                                   // count=1 (valid user descriptor)
    g0[1] = lds_byte_off;                         // lds_addr (bytes)
    g0[2] = (unsigned)(ga & 0xFFFFFFFFu);         // global_addr[31:0]
    g0[3] = (unsigned)((ga >> 32) & 0x1FFFFFFu) | (2u << 30);  // addr[56:32] | type=2

    const unsigned long long q0 =
        (1ull << 16) |                            // data_size = 2 bytes
        ((unsigned long long)(TILE_ELEMS & 0xFFFF) << 48);  // tensor_dim0[15:0]
    const unsigned long long q1 =
        ((unsigned long long)(TILE_ELEMS >> 16) & 0xFFFF) | // tensor_dim0[31:16]
        (1ull << 16) |                            // tensor_dim1 = 1
        ((unsigned long long)TILE_ELEMS << 48);   // tile_dim0 = 2048
    const unsigned long long q2 =
        ((unsigned long long)TILE_ELEMS << 32);   // tensor_dim0_stride[31:0]
    const unsigned long long q3 = 0ull;

    i32x8 g1;
    g1[0] = (int)q0; g1[1] = (int)(q0 >> 32);
    g1[2] = (int)q1; g1[3] = (int)(q1 >> 32);
    g1[4] = (int)q2; g1[5] = (int)(q2 >> 32);
    g1[6] = (int)q3; g1[7] = (int)(q3 >> 32);
    const i32x4 z4 = {0, 0, 0, 0};
    const i32x8 z8 = {0, 0, 0, 0, 0, 0, 0, 0};
    __builtin_amdgcn_tensor_load_to_lds(g0, g1, z4, z4, z8, 0);
}

// ---------------------------------------------------------------------------
// Kernel 0: zero output (atomic-accumulated) and expert counters.
// ---------------------------------------------------------------------------
__global__ void moe_zero_kernel(float* __restrict__ out, int n, int* __restrict__ cnt) {
    int i = blockIdx.x * blockDim.x + threadIdx.x;
    if (i < n) out[i] = 0.0f;
    if (i < E) cnt[i] = 0;
}

// ---------------------------------------------------------------------------
// Kernel 1: router (one wave per token) + x -> bf16 conversion.
// ---------------------------------------------------------------------------
__global__ __launch_bounds__(256)
void moe_router_kernel(const float* __restrict__ x, const float* __restrict__ Wg,
                       unsigned short* __restrict__ xb,
                       int* __restrict__ cnt, int* __restrict__ tok, float* __restrict__ wt) {
    const int wave = threadIdx.x >> 5;
    const int lane = threadIdx.x & 31;
    const int t = blockIdx.x * 8 + wave;
    if (t >= T) return;

    const float* xr = x + (size_t)t * D;
    float acc[E];
#pragma unroll
    for (int e = 0; e < E; ++e) acc[e] = 0.0f;

    for (int d = lane * 4; d < D; d += 128) {
        const float4 xv = *(const float4*)(xr + d);
        unsigned short* xo = xb + (size_t)t * D + d;
        xo[0] = f2bf(xv.x); xo[1] = f2bf(xv.y);
        xo[2] = f2bf(xv.z); xo[3] = f2bf(xv.w);
#pragma unroll
        for (int e = 0; e < E; ++e) {
            const float4 wv = *(const float4*)(Wg + (size_t)e * D + d);
            acc[e] += xv.x * wv.x + xv.y * wv.y + xv.z * wv.z + xv.w * wv.w;
        }
    }
#pragma unroll
    for (int e = 0; e < E; ++e)
#pragma unroll
        for (int off = 16; off > 0; off >>= 1)
            acc[e] += __shfl_xor(acc[e], off, 32);

    if (lane == 0) {
        int e1 = 0;
#pragma unroll
        for (int e = 1; e < E; ++e) if (acc[e] > acc[e1]) e1 = e;
        int e2 = (e1 == 0) ? 1 : 0;
#pragma unroll
        for (int e = 0; e < E; ++e) if (e != e1 && acc[e] > acc[e2]) e2 = e;
        const float m  = fmaxf(acc[e1], acc[e2]);
        const float p1 = __expf(acc[e1] - m);
        const float p2 = __expf(acc[e2] - m);
        const float s  = p1 + p2;
        int s1 = atomicAdd(&cnt[e1], 1);
        tok[e1 * T + s1] = t;  wt[e1 * T + s1] = p1 / s;
        int s2 = atomicAdd(&cnt[e2], 1);
        tok[e2 * T + s2] = t;  wt[e2 * T + s2] = p2 / s;
    }
}

// ---------------------------------------------------------------------------
// Kernel 2: repack weights fp32 -> bf16 in WMMA B-operand tile order.
// One block = one 32K x 64N tile -> 2048 contiguous bf16 (4KB), TDM-ready.
// Packed element order: ((n/16)*32 + (((k/8)%2)*16 | (n%16)))*16 + (k/16)*8 + (k%8)
// ---------------------------------------------------------------------------
__global__ __launch_bounds__(128)
void moe_repack_kernel(const float* __restrict__ w1, const float* __restrict__ w3,
                       const float* __restrict__ w2,
                       unsigned short* __restrict__ w1p, unsigned short* __restrict__ w3p,
                       unsigned short* __restrict__ w2p) {
    const int bid   = blockIdx.x;
    const int which = bid / (E * 1024);
    const int rem   = bid % (E * 1024);
    const int e     = rem / 1024;
    const int tile  = rem % 1024;

    const float* src; unsigned short* dst; int C;
    if (which == 0)      { src = w1; dst = w1p; C = F; }
    else if (which == 1) { src = w3; dst = w3p; C = F; }
    else                 { src = w2; dst = w2p; C = D; }
    const int ntn = C / 64;
    const int kt = tile / ntn, nt = tile % ntn;

    __shared__ float sT[32][65];
    const float* s0 = src + (size_t)e * D * F + (size_t)(kt * 32) * C + nt * 64;
    for (int i = threadIdx.x; i < 32 * 16; i += 128) {
        const int k = i >> 4, n4 = (i & 15) * 4;
        const float4 v = *(const float4*)(s0 + (size_t)k * C + n4);
        sT[k][n4 + 0] = v.x; sT[k][n4 + 1] = v.y;
        sT[k][n4 + 2] = v.z; sT[k][n4 + 3] = v.w;
    }
    __syncthreads();

    const int tid = threadIdx.x;
    const int sub = tid >> 5, l = tid & 31;
    const int n  = sub * 16 + (l & 15);
    const int kb = (l >> 4) << 4;
    unsigned short outv[16];
#pragma unroll
    for (int j = 0; j < 16; ++j) outv[j] = f2bf(sT[(kb >> 1) + (j & 7) + ((j >> 3) << 4)][n]);
    uint4* dv = (uint4*)(dst + ((size_t)(e * 1024 + tile)) * TILE_ELEMS + tid * 16);
    dv[0] = *(const uint4*)(outv);
    dv[1] = *(const uint4*)(outv + 8);
}

// ---------------------------------------------------------------------------
// Kernel 3: grouped expert GEMM, TDM-fed, double-buffered.
//   A operand: 2x global_load_b128 per lane straight from bf16 xb (gathered).
//   B operand: tensor_load_to_lds of pre-packed 4KB tiles, s_wait_tensorcnt.
// ---------------------------------------------------------------------------
__global__ __launch_bounds__(128)
void moe_expert_kernel(const unsigned short* __restrict__ xb,
                       const unsigned short* __restrict__ w1p,
                       const unsigned short* __restrict__ w3p,
                       const unsigned short* __restrict__ w2p,
                       const int* __restrict__ cnt, const int* __restrict__ tok,
                       const float* __restrict__ wt, float* __restrict__ out) {
    const int e    = blockIdx.x / MAXTILE;
    const int tile = blockIdx.x % MAXTILE;
    const int n_e  = cnt[e];
    if (tile * TM >= n_e) return;

    extern __shared__ char smem[];
    unsigned short* sB1  = (unsigned short*)(smem + LDS_B1);
    unsigned short* sB2  = (unsigned short*)(smem + LDS_B2);
    unsigned short* sAct = (unsigned short*)(smem + LDS_ACT);
    int*   sTok = (int*)(smem + LDS_TOK);
    float* sWt  = (float*)(smem + LDS_WT);
    const unsigned ldsBase = __builtin_amdgcn_groupstaticsize();

    const int tid  = threadIdx.x;
    const int lane = tid & 31;
    const int wave = tid >> 5;

    if (tid < TM) {
        const int r = tile * TM + tid;
        const bool valid = r < n_e;
        sTok[tid] = valid ? tok[e * T + r] : 0;
        sWt[tid]  = valid ? wt[e * T + r] : 0.0f;
    }
    __syncthreads();

    // Per-lane A fragment source: row m = lane%16, K-half kh = lane/16.
    const unsigned short* arow = xb + (size_t)sTok[lane & 15] * D;
    const int kh = lane >> 4;
    union ABu { v16bf v; v8bf h[2]; };

    // ---------------- Phase A: act = silu(X W1) * (X W3) ----------------
    for (int f0 = 0; f0 < F; f0 += BN) {
        v8f accg = {}, accu = {};
        int cur = 0;
        const size_t tcol = (size_t)(f0 >> 6);
        if (tid < 32) {
            const size_t t0 = ((size_t)(e * (D / 32)) * (F / 64) + tcol) * TILE_ELEMS;
            tdm_load_tile(ldsBase + LDS_B1, w1p + t0);
            tdm_load_tile(ldsBase + LDS_B2, w3p + t0);
        }
        for (int d0 = 0; d0 < D; d0 += KK) {
            const bool more = (d0 + KK) < D;
            if (tid < 32) {
                if (more) {
                    const size_t tn = ((size_t)(e * (D / 32) + ((d0 + KK) >> 5)) * (F / 64)
                                       + tcol) * TILE_ELEMS;
                    tdm_load_tile(ldsBase + LDS_B1 + (cur ^ 1) * TILE_BYTES, w1p + tn);
                    tdm_load_tile(ldsBase + LDS_B2 + (cur ^ 1) * TILE_BYTES, w3p + tn);
                    __builtin_amdgcn_s_wait_tensorcnt((short)2);
                } else {
                    __builtin_amdgcn_s_wait_tensorcnt((short)0);
                }
            }
            __syncthreads();
            ABu a;
            a.h[0] = *(const v8bf*)(arow + d0 + 8 * kh);
            a.h[1] = *(const v8bf*)(arow + d0 + 16 + 8 * kh);
            const v16bf b1 = *(const v16bf*)(sB1 + cur * TILE_ELEMS + (wave * 32 + lane) * 16);
            const v16bf b2 = *(const v16bf*)(sB2 + cur * TILE_ELEMS + (wave * 32 + lane) * 16);
            accg = wmma_bf16(a.v, b1, accg);
            accu = wmma_bf16(a.v, b2, accu);
            __syncthreads();
            cur ^= 1;
        }
        // SwiGLU; store into sAct in packed-A chunk layout for Phase B.
        const int f_n = f0 + wave * 16 + (lane & 15);
        const int kh2 = lane >> 4;
#pragma unroll
        for (int r = 0; r < 8; ++r) {
            const int m = r + 8 * kh2;
            const float g = accg[r];
            const float u = accu[r];
            const float av = (g / (1.0f + __expf(-g))) * u;
            const int c   = f_n >> 5;
            const int kl  = f_n & 31;
            const int khp = (kl >> 3) & 1;
            const int jp  = (kl & 7) + ((kl >> 4) << 3);
            sAct[(((c * 32) + (m & 15) + 16 * khp) * 16) + jp] = f2bf(av);
        }
    }
    __syncthreads();

    // ---------------- Phase B: out += weight * (act W2) ----------------
    for (int n0 = 0; n0 < D; n0 += BN) {
        v8f acc = {};
        int cur = 0;
        const size_t ncol = (size_t)(n0 >> 6);
        if (tid < 32) {
            const size_t t0 = ((size_t)(e * (F / 32)) * (D / 64) + ncol) * TILE_ELEMS;
            tdm_load_tile(ldsBase + LDS_B1, w2p + t0);
        }
        for (int k0 = 0; k0 < F; k0 += KK) {
            const bool more = (k0 + KK) < F;
            if (tid < 32) {
                if (more) {
                    const size_t tn = ((size_t)(e * (F / 32) + ((k0 + KK) >> 5)) * (D / 64)
                                       + ncol) * TILE_ELEMS;
                    tdm_load_tile(ldsBase + LDS_B1 + (cur ^ 1) * TILE_BYTES, w2p + tn);
                    __builtin_amdgcn_s_wait_tensorcnt((short)1);
                } else {
                    __builtin_amdgcn_s_wait_tensorcnt((short)0);
                }
            }
            __syncthreads();
            const v16bf a = *(const v16bf*)(sAct + (((k0 >> 5) * 32 + lane) * 16));
            const v16bf b = *(const v16bf*)(sB1 + cur * TILE_ELEMS + (wave * 32 + lane) * 16);
            acc = wmma_bf16(a, b, acc);
            __syncthreads();
            cur ^= 1;
        }
        const int nc  = n0 + wave * 16 + (lane & 15);
        const int kh2 = lane >> 4;
#pragma unroll
        for (int r = 0; r < 8; ++r) {
            const int m = r + 8 * kh2;
            if (tile * TM + m < n_e) {
                atomicAdd(&out[(size_t)sTok[m] * D + nc], acc[r] * sWt[m]);
            }
        }
    }
}

// ---------------------------------------------------------------------------
extern "C" void kernel_launch(void* const* d_in, const int* in_sizes, int n_in,
                              void* d_out, int out_size, void* d_ws, size_t ws_size,
                              hipStream_t stream) {
    (void)in_sizes; (void)n_in; (void)out_size; (void)ws_size;
    const float* x  = (const float*)d_in[0];
    const float* Wg = (const float*)d_in[1];
    const float* w1 = (const float*)d_in[2];
    const float* w3 = (const float*)d_in[3];
    const float* w2 = (const float*)d_in[4];
    float* out = (float*)d_out;

    // Workspace layout
    char* p = (char*)d_ws;
    unsigned short* xb  = (unsigned short*)p;  p += (size_t)T * D * 2;      // 16.8 MB
    int*   cnt = (int*)p;                      p += 256;
    int*   tok = (int*)p;                      p += (size_t)E * T * 4;
    float* wt  = (float*)p;                    p += (size_t)E * T * 4;
    unsigned short* w1p = (unsigned short*)p;  p += (size_t)E * D * F * 2;  // 33.5 MB
    unsigned short* w3p = (unsigned short*)p;  p += (size_t)E * D * F * 2;
    unsigned short* w2p = (unsigned short*)p;

    // 0) zero output + counters
    moe_zero_kernel<<<(T * D + 255) / 256, 256, 0, stream>>>(out, T * D, cnt);
    // 1) router + bf16 conversion of activations
    moe_router_kernel<<<T / 8, 256, 0, stream>>>(x, Wg, xb, cnt, tok, wt);
    // 2) weight repack into WMMA-ready bf16 tiles (one-time, ~12us at HBM)
    moe_repack_kernel<<<3 * E * 1024, 128, 0, stream>>>(w1, w3, w2, w1p, w3p, w2p);
    // 3) grouped expert GEMM, TDM-fed
    moe_expert_kernel<<<E * MAXTILE, 128, LDS_TOTAL, stream>>>(
        xb, w1p, w3p, w2p, cnt, tok, wt, out);
}
